// GATv2Layer_46514495816135
// MI455X (gfx1250) — compile-verified, matrix operand order checked by reading, not run on hardware
//
#include <hip/hip_runtime.h>
#include <hip/hip_bf16.h>
#include <math.h>

#define N_NODES 100000
#define N_EDGES 1600000
#define HC      128
#define NHEAD   4
#define NEG_SLOPE 0.2f

typedef __attribute__((ext_vector_type(16))) __bf16 v16bf;
typedef __attribute__((ext_vector_type(8)))  float  v8f;

__device__ __forceinline__ void atomic_max_f32(float* addr, float val) {
    // int-punned float max: valid mixed pos/neg when initialized to -inf
    if (val >= 0.0f) atomicMax((int*)addr, __float_as_int(val));
    else             atomicMin((unsigned int*)addr, __float_as_uint(val));
}

// ---------------- init: zero accumulator/denom, -inf seg_max ----------------
__global__ void k_init(float* __restrict__ out, float* __restrict__ seg_max,
                       float* __restrict__ denom) {
    int i = blockIdx.x * blockDim.x + threadIdx.x;
    if (i < N_NODES * HC) out[i] = 0.0f;
    if (i < N_NODES * NHEAD) {
        seg_max[i] = -__builtin_huge_valf();
        denom[i]   = 0.0f;
    }
}

// ---- swizzle W_l|W_r into per-lane WMMA B-fragment records (32B each) ------
// Record index r = ((mat*8 + tn)*4 + kk)*32 + lane. Record contents (16 bf16):
//   j in 0..7  -> W[kb + j     ][n]   with kb = kk*32 + (lane>>4)*8
//   j in 8..15 -> W[kb + 16 + j-8][n]      n  = tn*16 + (lane&15)
// matching the ISA 16-bit 32x16 B-matrix VGPR layout.
__global__ void k_swizzle_w(const float* __restrict__ Wl,
                            const float* __restrict__ Wr,
                            __bf16* __restrict__ wfrag) {
    const int t = blockIdx.x * blockDim.x + threadIdx.x;   // 0..2047
    if (t >= 2048) return;
    const int lane = t & 31;
    const int kk   = (t >> 5) & 3;
    const int tn   = (t >> 7) & 7;
    const int mat  = (t >> 10) & 1;
    const float* W = mat ? Wr : Wl;
    const int n  = tn * 16 + (lane & 15);
    const int kb = kk * 32 + (lane >> 4) * 8;
    __bf16* o = wfrag + (size_t)t * 16;
#pragma unroll
    for (int j = 0; j < 8; ++j) {
        o[j]     = (__bf16)W[(kb + j)      * 128 + n];
        o[8 + j] = (__bf16)W[(kb + 16 + j) * 128 + n];
    }
}

// ---------------- fused xl = xW_l+b_l, xr = xW_r+b_r via WMMA ----------------
// block = 256 threads = 8 waves; wave computes 16 rows x 128 cols for BOTH W.
// Pre-swizzled weight fragments staged to LDS via async global->LDS copies.
// N_NODES % 16 == 0, so any live wave covers fully valid rows: no store guards.
__global__ __launch_bounds__(256) void k_gemm_wmma(
        const float* __restrict__ x, const __bf16* __restrict__ wfrag,
        const float* __restrict__ b_l, const float* __restrict__ b_r,
        float* __restrict__ xl, float* __restrict__ xr) {
    __shared__ __bf16 lds_frag[2 * 128 * 128];   // 64 KB
    {
        // 4096 x 16B chunks; each thread issues 16 async 128b LDS loads.
        // LDS byte address = low 32 bits of the generic pointer (aperture is in
        // the high dword). Passing lds_frag into the asm makes the write visible
        // to the compiler so the later ds_load fragment reads are not folded.
        const unsigned tid = threadIdx.x;
        const unsigned ldsbase = (unsigned)(size_t)(void*)lds_frag;
#pragma unroll
        for (int i = 0; i < 16; ++i) {
            unsigned off = (tid + i * 256u) * 16u;
            asm volatile("global_load_async_to_lds_b128 %0, %1, %2"
                         :
                         : "v"(ldsbase + off), "v"(off), "s"(wfrag), "r"(lds_frag)
                         : "memory");
        }
        asm volatile("s_wait_asynccnt 0" ::: "memory");
    }
    __syncthreads();

    const int wave = threadIdx.x >> 5;
    const int lane = threadIdx.x & 31;
    const int m0 = blockIdx.x * 128 + wave * 16;
    if (m0 >= N_NODES) return;                 // wave-uniform guard

    const int mrow = m0 + (lane & 15);
    const int hi8 = (lane >> 4) * 8;           // K sub-offset for hi half-wave
    const int nl  = lane & 15;

    // A fragments: 16x32 bf16 tiles for K-steps 0..3, converted in registers.
    // ISA layout: lane m (0-15), halves 0-7 = K kb..kb+7, halves 8-15 = K kb+16..kb+23
    v16bf afrag[4];
    const float* xrow = x + (size_t)mrow * 128;
#pragma unroll
    for (int kk = 0; kk < 4; ++kk) {
        const int kb = kk * 32 + hi8;
        float4 p0 = *(const float4*)(xrow + kb);
        float4 p1 = *(const float4*)(xrow + kb + 4);
        float4 p2 = *(const float4*)(xrow + kb + 16);
        float4 p3 = *(const float4*)(xrow + kb + 20);
        afrag[kk][0]  = (__bf16)p0.x; afrag[kk][1]  = (__bf16)p0.y;
        afrag[kk][2]  = (__bf16)p0.z; afrag[kk][3]  = (__bf16)p0.w;
        afrag[kk][4]  = (__bf16)p1.x; afrag[kk][5]  = (__bf16)p1.y;
        afrag[kk][6]  = (__bf16)p1.z; afrag[kk][7]  = (__bf16)p1.w;
        afrag[kk][8]  = (__bf16)p2.x; afrag[kk][9]  = (__bf16)p2.y;
        afrag[kk][10] = (__bf16)p2.z; afrag[kk][11] = (__bf16)p2.w;
        afrag[kk][12] = (__bf16)p3.x; afrag[kk][13] = (__bf16)p3.y;
        afrag[kk][14] = (__bf16)p3.z; afrag[kk][15] = (__bf16)p3.w;
    }

    // Per-lane fragment records: 32B contiguous -> ds_load_b128 pairs per frag.
    const char* myfrag = (const char*)lds_frag + (size_t)lane * 32;

#pragma unroll
    for (int t = 0; t < 16; ++t) {             // 8 n-tiles x {W_l, W_r}
        const int mat = t >> 3;
        const int ng  = (t & 7) * 16 + nl;     // global output column
        const float bval = (mat ? b_r : b_l)[ng];
        v8f acc;
#pragma unroll
        for (int r = 0; r < 8; ++r) acc[r] = bval;

#pragma unroll
        for (int kk = 0; kk < 4; ++kk) {
            v16bf bfrag = *(const v16bf*)(myfrag + (size_t)(t * 4 + kk) * 1024);
            acc = __builtin_amdgcn_wmma_f32_16x16x32_bf16(
                      false, afrag[kk], false, bfrag, (short)0, acc, false, false);
        }
        float* dstp = (mat ? xr : xl) + (size_t)(m0 + hi8) * 128 + ng;
#pragma unroll
        for (int r = 0; r < 8; ++r) {          // D: VGPR r -> row m0 + hi8 + r
            dstp[(size_t)r * 128] = acc[r];
        }
    }
}

// ---------------- edge pass 1: logits + segment max (wave per edge) ---------
__global__ __launch_bounds__(256) void k_edge_logits(
        const float* __restrict__ xl, const float* __restrict__ xr,
        const int* __restrict__ src, const int* __restrict__ dst,
        const float* __restrict__ ew, const float* __restrict__ We,
        const float* __restrict__ att, float* __restrict__ logits,
        float* __restrict__ seg_max) {
    const int gw = (blockIdx.x * blockDim.x + threadIdx.x) >> 5;
    const int lane = threadIdx.x & 31;
    if (gw >= N_EDGES) return;
    const int s = src[gw], d = dst[gw];
    const float w = ew[gw];
    float4 a  = ((const float4*)xl)[(size_t)s * 32 + lane];
    float4 b  = ((const float4*)xr)[(size_t)d * 32 + lane];
    float4 ef = ((const float4*)We)[lane];
    float4 g;
    g.x = a.x + b.x + w * ef.x;  g.y = a.y + b.y + w * ef.y;
    g.z = a.z + b.z + w * ef.z;  g.w = a.w + b.w + w * ef.w;
    g.x = g.x >= 0.0f ? g.x : NEG_SLOPE * g.x;
    g.y = g.y >= 0.0f ? g.y : NEG_SLOPE * g.y;
    g.z = g.z >= 0.0f ? g.z : NEG_SLOPE * g.z;
    g.w = g.w >= 0.0f ? g.w : NEG_SLOPE * g.w;
    float4 at = ((const float4*)att)[lane];
    float p = g.x * at.x + g.y * at.y + g.z * at.z + g.w * at.w;
    // reduce over 8-lane head group (head h = lane>>3 covers C=32 = 8 lanes)
    p += __shfl_xor(p, 1, 32);
    p += __shfl_xor(p, 2, 32);
    p += __shfl_xor(p, 4, 32);
    if ((lane & 7) == 0) {
        const int h = lane >> 3;
        logits[(size_t)gw * 4 + h] = p;
        atomic_max_f32(&seg_max[(size_t)d * 4 + h], p);
    }
}

// ---------------- edge pass 2: exp + segment sum (thread per edge*head) -----
__global__ void k_edge_exp(const int* __restrict__ dst, float* __restrict__ logits,
                           const float* __restrict__ seg_max, float* __restrict__ denom) {
    const int t = blockIdx.x * blockDim.x + threadIdx.x;
    if (t >= N_EDGES * NHEAD) return;
    const int e = t >> 2, h = t & 3;
    const int d = dst[e];
    const float ex = __expf(logits[t] - seg_max[(size_t)d * 4 + h]);
    logits[t] = ex;                          // reuse buffer for numerator
    atomicAdd(&denom[(size_t)d * 4 + h], ex);
}

// ---------------- edge pass 3: alpha-weighted scatter (wave per edge) -------
__global__ __launch_bounds__(256) void k_edge_scatter(
        const float* __restrict__ xl, const int* __restrict__ src,
        const int* __restrict__ dst, const float* __restrict__ ex,
        const float* __restrict__ denom, float* __restrict__ out) {
    const int gw = (blockIdx.x * blockDim.x + threadIdx.x) >> 5;
    const int lane = threadIdx.x & 31;
    if (gw >= N_EDGES) return;
    const int s = src[gw], d = dst[gw];
    const int h = lane >> 3;
    const float alpha = ex[(size_t)gw * 4 + h] / denom[(size_t)d * 4 + h];
    float4 m = ((const float4*)xl)[(size_t)s * 32 + lane];
    float* o = out + (size_t)d * 128 + lane * 4;
    atomicAdd(o + 0, alpha * m.x);
    atomicAdd(o + 1, alpha * m.y);
    atomicAdd(o + 2, alpha * m.z);
    atomicAdd(o + 3, alpha * m.w);
}

// ---------------- epilogue: +bias, ELU, LayerNorm (wave per row) ------------
__global__ __launch_bounds__(256) void k_finalize(
        float* __restrict__ out, const float* __restrict__ bias,
        const float* __restrict__ gamma, const float* __restrict__ beta) {
    const int gw = (blockIdx.x * blockDim.x + threadIdx.x) >> 5;
    const int lane = threadIdx.x & 31;
    if (gw >= N_NODES) return;
    float4* row = (float4*)(out + (size_t)gw * 128);
    float4 v = row[lane];
    float4 b = ((const float4*)bias)[lane];
    v.x += b.x; v.y += b.y; v.z += b.z; v.w += b.w;
    v.x = v.x > 0.0f ? v.x : expm1f(v.x);
    v.y = v.y > 0.0f ? v.y : expm1f(v.y);
    v.z = v.z > 0.0f ? v.z : expm1f(v.z);
    v.w = v.w > 0.0f ? v.w : expm1f(v.w);
    float s = v.x + v.y + v.z + v.w;
    float q = v.x * v.x + v.y * v.y + v.z * v.z + v.w * v.w;
#pragma unroll
    for (int m = 1; m < 32; m <<= 1) {
        s += __shfl_xor(s, m, 32);
        q += __shfl_xor(q, m, 32);
    }
    const float mu  = s * (1.0f / 128.0f);
    const float var = q * (1.0f / 128.0f) - mu * mu;
    const float sc  = rsqrtf(var + 1e-5f);
    float4 g  = ((const float4*)gamma)[lane];
    float4 be = ((const float4*)beta)[lane];
    v.x = (v.x - mu) * sc * g.x + be.x;
    v.y = (v.y - mu) * sc * g.y + be.y;
    v.z = (v.z - mu) * sc * g.z + be.z;
    v.w = (v.w - mu) * sc * g.w + be.w;
    row[lane] = v;
}

extern "C" void kernel_launch(void* const* d_in, const int* in_sizes, int n_in,
                              void* d_out, int out_size, void* d_ws, size_t ws_size,
                              hipStream_t stream) {
    const float* x    = (const float*)d_in[0];
    const int*   ei   = (const int*)  d_in[1];
    const float* ew   = (const float*)d_in[2];
    const float* W_l  = (const float*)d_in[3];
    const float* b_l  = (const float*)d_in[4];
    const float* W_r  = (const float*)d_in[5];
    const float* b_r  = (const float*)d_in[6];
    const float* W_e  = (const float*)d_in[7];
    const float* att  = (const float*)d_in[8];
    const float* bias = (const float*)d_in[9];
    const float* gam  = (const float*)d_in[10];
    const float* bet  = (const float*)d_in[11];
    float* out = (float*)d_out;

    char* ws = (char*)d_ws;
    __bf16* wfrag  = (__bf16*)ws;                          // 64 KB (swizzled)
    float*  xl     = (float*)(ws + 65536);                 // 51.2 MB
    float*  xr     = xl + (size_t)N_NODES * 128;           // 51.2 MB
    float*  logits = xr + (size_t)N_NODES * 128;           // 25.6 MB
    float*  smax   = logits + (size_t)N_EDGES * 4;         // 1.6 MB
    float*  denom  = smax + (size_t)N_NODES * 4;           // 1.6 MB

    const int* src = ei;
    const int* dst = ei + N_EDGES;

    k_init<<<(N_NODES * 128 + 255) / 256, 256, 0, stream>>>(out, smax, denom);
    k_swizzle_w<<<8, 256, 0, stream>>>(W_l, W_r, wfrag);
    k_gemm_wmma<<<(N_NODES + 127) / 128, 256, 0, stream>>>(x, wfrag, b_l, b_r, xl, xr);
    k_edge_logits<<<(N_EDGES + 7) / 8, 256, 0, stream>>>(xl, xr, src, dst, ew, W_e,
                                                         att, logits, smax);
    k_edge_exp<<<(N_EDGES * 4 + 255) / 256, 256, 0, stream>>>(dst, logits, smax, denom);
    k_edge_scatter<<<(N_EDGES + 7) / 8, 256, 0, stream>>>(xl, src, dst, logits, denom, out);
    k_finalize<<<(N_NODES + 7) / 8, 256, 0, stream>>>(out, bias, gam, bet);
}
